// ECST_85856396247628
// MI455X (gfx1250) — compile-verified, matrix operand order checked by reading, not run on hardware
//
#include <hip/hip_runtime.h>
#include <hip/hip_bf16.h>

// ---------------------------------------------------------------------------
// Problem constants (from the reference)
// ---------------------------------------------------------------------------
#define BATCH    2048
#define NUM_ENT  100000
#define D        128
#define ND       32
#define NNB      10
#define THRESH   1373

typedef __attribute__((ext_vector_type(16))) __bf16 v16bf;
typedef __attribute__((ext_vector_type(8)))  float  v8f;

union Frag16 {             // one 16x32-bf16 A or 32x16-bf16 B fragment per lane
    v16bf v;
    uint4 q[2];            // two contiguous 16-byte halves
};

// ---------------------------------------------------------------------------
// Kernel 0: fp32 -> bf16 conversion of ent_embed (12.8M elems), vectorized x4
// ---------------------------------------------------------------------------
__global__ void __launch_bounds__(256)
cvt_f32_to_bf16(const float* __restrict__ in, __hip_bfloat16* __restrict__ out, int n4) {
    int i = blockIdx.x * blockDim.x + threadIdx.x;
    if (i >= n4) return;
    float4 v = *((const float4*)in + i);
    union { __hip_bfloat16 h[4]; uint2 u; } o;
    o.h[0] = __float2bfloat16(v.x);
    o.h[1] = __float2bfloat16(v.y);
    o.h[2] = __float2bfloat16(v.z);
    o.h[3] = __float2bfloat16(v.w);
    *((uint2*)out + i) = o.u;
}

// ---------------------------------------------------------------------------
// Kernel 1: head stage. One 128-thread block per batch row.
// att==1 exactly (softmax over singleton axis) => agg = W_V * (sum_n tn) + 10*b_V.
// Q/K path is dead code and omitted. Emits F as bf16 for the tail GEMM.
// ---------------------------------------------------------------------------
__global__ void __launch_bounds__(128)
head_stage(const int* __restrict__ src, const int* __restrict__ rel,
           const int* __restrict__ t_idxs,
           const float* __restrict__ ent, const float* __restrict__ rel_emb,
           const float* __restrict__ nod,
           const float* __restrict__ W_V, const float* __restrict__ b_V,
           const float* __restrict__ fc1_w, const float* __restrict__ fc1_b,
           const float* __restrict__ fc2_w, const float* __restrict__ fc2_b,
           __hip_bfloat16* __restrict__ F_out) {
    __shared__ int   nidx[NNB];
    __shared__ float tn[D + ND];    // 160
    __shared__ float z[2 * D];      // 256
    __shared__ float hbuf[D];       // 128

    const int b = blockIdx.x;
    const int t = threadIdx.x;      // 0..127
    const int s = src[b];

    if (t < NNB) nidx[t] = t_idxs[s * NNB + t];
    __syncthreads();

    // tn_sum[0:128] = sum_n ent_embed[neigh]; tn_sum[128:160] = sum_n nod_embed[flag]
    float es = 0.f;
    #pragma unroll
    for (int n = 0; n < NNB; ++n)
        es += ent[(size_t)nidx[n] * D + t];
    tn[t] = es;
    if (t < ND) {
        int c1 = 0;
        #pragma unroll
        for (int n = 0; n < NNB; ++n) c1 += (nidx[n] >= THRESH) ? 1 : 0;
        tn[D + t] = (float)(NNB - c1) * nod[t] + (float)c1 * nod[ND + t];
    }
    __syncthreads();

    // V_src = ent[src] + W_V @ tn_sum + NNB*b_V ; z = [V_src, rel_embed[rel]]
    float agg = (float)NNB * b_V[t];
    #pragma unroll 4
    for (int j = 0; j < D + ND; ++j) agg += W_V[t * (D + ND) + j] * tn[j];
    z[t]     = ent[(size_t)s * D + t] + agg;
    z[D + t] = rel_emb[rel[b] * D + t];
    __syncthreads();

    float hv = fc1_b[t];
    #pragma unroll 4
    for (int j = 0; j < 2 * D; ++j) hv += fc1_w[t * 2 * D + j] * z[j];
    hbuf[t] = fmaxf(hv, 0.f);
    __syncthreads();

    float f = fc2_b[t];
    #pragma unroll 4
    for (int j = 0; j < D; ++j) f += fc2_w[t * D + j] * hbuf[j];
    F_out[(size_t)b * D + t] = __float2bfloat16(f);
}

// ---------------------------------------------------------------------------
// Kernel 2: out[2048, 100000] = sigmoid(F @ E^T), bf16 WMMA, fp32 accumulate.
// One WG per 64-entity slab (E slab staged in 16KB LDS). 8 waves sweep all
// 128 M-tiles; each wave: 16x64 strip = 4 nsub x 4 ksteps of
// v_wmma_f32_16x16x32_bf16, fused sigmoid epilogue.
// ---------------------------------------------------------------------------
__global__ void __launch_bounds__(256)
gemm_sigmoid(const __hip_bfloat16* __restrict__ Fb,   // [2048,128] bf16
             const __hip_bfloat16* __restrict__ Eb,   // [100000,128] bf16
             float* __restrict__ out) {               // [2048,100000]
    __shared__ uint4 ldsE[64 * 16];   // 64 rows x 128 bf16 = 16KB

    const int nbase = blockIdx.x * 64;
    const int tid   = threadIdx.x;
    const int wave  = tid >> 5;       // 0..7
    const int lane  = tid & 31;
    const int lrow  = lane & 15;      // M (A) / N (B) index within tile
    const int khalf = lane >> 4;      // which K-half this lane holds

    // Stage E slab: 1024 uint4, 4 per thread; clamp OOB rows to 0 (stores guarded).
    #pragma unroll
    for (int i = tid; i < 64 * 16; i += 256) {
        int row = i >> 4;
        int c   = i & 15;
        int e   = nbase + row;
        if (e >= NUM_ENT) e = 0;
        ldsE[i] = *((const uint4*)(Eb + (size_t)e * D) + c);
    }
    __syncthreads();

    for (int mt = wave; mt < BATCH / 16; mt += 8) {
        // --- A fragments: 16x32 bf16 per ktile, ISA lane layout ---
        const int m = mt * 16 + lrow;
        const uint4* arow = (const uint4*)(Fb + (size_t)m * D);
        Frag16 a[4];
        #pragma unroll
        for (int kt = 0; kt < 4; ++kt) {
            a[kt].q[0] = arow[kt * 4 + khalf];        // K = kt*32 + khalf*8 ..
            a[kt].q[1] = arow[kt * 4 + khalf + 2];    // K = kt*32 + 16 + khalf*8 ..
        }

        v8f acc[4];
        #pragma unroll
        for (int ns = 0; ns < 4; ++ns) acc[ns] = (v8f){0.f,0.f,0.f,0.f,0.f,0.f,0.f,0.f};

        #pragma unroll
        for (int ns = 0; ns < 4; ++ns) {
            const int brow = ns * 16 + lrow;          // entity column within slab
            #pragma unroll
            for (int kt = 0; kt < 4; ++kt) {
                Frag16 bfrag;
                bfrag.q[0] = ldsE[brow * 16 + kt * 4 + khalf];
                bfrag.q[1] = ldsE[brow * 16 + kt * 4 + khalf + 2];
                acc[ns] = __builtin_amdgcn_wmma_f32_16x16x32_bf16(
                    false, a[kt].v, false, bfrag.v, (short)0, acc[ns], false, false);
            }
        }

        // --- epilogue: sigmoid + store (C/D layout: VGPR i -> M=i+8*(lane/16), N=lane%16)
        const int mrow0 = mt * 16 + khalf * 8;
        #pragma unroll
        for (int ns = 0; ns < 4; ++ns) {
            const int n = nbase + ns * 16 + lrow;
            if (n < NUM_ENT) {
                #pragma unroll
                for (int i = 0; i < 8; ++i) {
                    float x = acc[ns][i];
                    out[(size_t)(mrow0 + i) * NUM_ENT + n] = 1.0f / (1.0f + __expf(-x));
                }
            }
        }
    }
}

// ---------------------------------------------------------------------------
// Launch
// ---------------------------------------------------------------------------
extern "C" void kernel_launch(void* const* d_in, const int* in_sizes, int n_in,
                              void* d_out, int out_size, void* d_ws, size_t ws_size,
                              hipStream_t stream) {
    const int*   src     = (const int*)  d_in[0];
    const int*   rel     = (const int*)  d_in[1];
    const int*   t_idxs  = (const int*)  d_in[2];
    const float* ent     = (const float*)d_in[3];
    const float* rel_emb = (const float*)d_in[4];
    const float* nod     = (const float*)d_in[5];
    // d_in[6..9] = W_Q, b_Q, W_K, b_K : provably dead (softmax over singleton axis)
    const float* W_V     = (const float*)d_in[10];
    const float* b_V     = (const float*)d_in[11];
    const float* fc1_w   = (const float*)d_in[12];
    const float* fc1_b   = (const float*)d_in[13];
    const float* fc2_w   = (const float*)d_in[14];
    const float* fc2_b   = (const float*)d_in[15];

    // workspace: [ E_bf16 : 100000*128*2B = 25.6MB ][ F_bf16 : 2048*128*2B ]
    __hip_bfloat16* Eb = (__hip_bfloat16*)d_ws;
    __hip_bfloat16* Fb = Eb + (size_t)NUM_ENT * D;

    const int n4 = (NUM_ENT * D) / 4;   // 3.2M float4 groups
    cvt_f32_to_bf16<<<(n4 + 255) / 256, 256, 0, stream>>>(ent, Eb, n4);

    head_stage<<<BATCH, 128, 0, stream>>>(src, rel, t_idxs, ent, rel_emb, nod,
                                          W_V, b_V, fc1_w, fc1_b, fc2_w, fc2_b, Fb);

    const int nblocks = (NUM_ENT + 63) / 64;   // 1563
    gemm_sigmoid<<<nblocks, 256, 0, stream>>>(Fb, Eb, (float*)d_out);
}
